// LSTMLayer_74783970558045
// MI455X (gfx1250) — matrix operand order repaired and checked by reading, not compiled, not run on hardware
//
#include <hip/hip_runtime.h>

// ---------------------------------------------------------------------------
// LSTM layer for MI455X (gfx1250), wave32 + v_wmma_f32_16x16x32_bf16.
// T=512, B=128, I=256, H=1024. Per timestep (512 sequential launches):
//   gates = [x_t | h_{t-1}] @ [W_ih | W_hh]^T + b_ih + b_hh
// computed as 40 K-slices of 32. B-operand (weight^T) fragments are staged
// in LDS (double-buffered, 8 KB) since all 8 waves of a block share them;
// A-operand (x/h rows) loaded directly per wave. f32 WMMA accumulation,
// f32 cell state; bf16 operand copies built once per launch in workspace.
// ---------------------------------------------------------------------------

typedef __attribute__((ext_vector_type(16))) __bf16 v16bf;
typedef __attribute__((ext_vector_type(8)))  __bf16 v8bf;
typedef __attribute__((ext_vector_type(8)))  float  v8f;

namespace {
constexpr int T = 512;
constexpr int B = 128;
constexpr int I = 256;
constexpr int H = 1024;
constexpr int KSLICES = (I + H) / 32;   // 8 input + 32 recurrent = 40
}

// ---- fp32 -> bf16 conversion (RNE via hardware cvt) ------------------------
__global__ void f32_to_bf16_kernel(const float* __restrict__ src,
                                   __bf16* __restrict__ dst, int n) {
    int i = blockIdx.x * blockDim.x + threadIdx.x;
    if (i < n) dst[i] = (__bf16)src[i];
}

// ---- WMMA A fragment: 16x32 bf16 (ISA 7.12.2) ------------------------------
// lane l<16: row M=l, K in {0..15}; lanes 16..31: same rows, K +16 (contiguous K chosen here, B matches).
__device__ inline v16bf load_a16x32(const __bf16* __restrict__ base, int ld,
                                    int rowBase, int k0, int lane) {
    const int r  = lane & 15;
    const int kb = (lane & 16) ? 16 : 0;
    const __bf16* p = base + (size_t)(rowBase + r) * ld + k0 + kb;
    v8bf lo = *reinterpret_cast<const v8bf*>(p);       // K = k0+kb   .. +7
    v8bf hi = *reinterpret_cast<const v8bf*>(p + 8);   // K = k0+8+kb .. +7
    return __builtin_shufflevector(lo, hi, 0, 1, 2, 3, 4, 5, 6, 7,
                                   8, 9, 10, 11, 12, 13, 14, 15);
}

__device__ inline float sigmoidf_fast(float x) {
    return 1.0f / (1.0f + __expf(-x));
}

// Per-slice operand source: slices 0..7 -> (x_t, W_ih), 8..39 -> (h, W_hh).
struct SliceSrc { const __bf16* A; const __bf16* W; int ld; int k0; };
__device__ inline SliceSrc slice_src(int s, const __bf16* xb, const __bf16* h_in,
                                     const __bf16* Wih, const __bf16* Whh) {
    SliceSrc r;
    if (s < I / 32) { r.A = xb;   r.W = Wih; r.ld = I; r.k0 = s * 32; }
    else            { r.A = h_in; r.W = Whh; r.ld = H; r.k0 = (s - I / 32) * 32; }
    return r;
}

// ---- one LSTM timestep -----------------------------------------------------
// grid = H/16 = 64 blocks; block = 256 threads = 8 waves.
// wave w: batch rows [16w,16w+16); block bx: hidden cols [16bx,16bx+16).
// Each wave accumulates 4 gate tiles (i,f,g,o) -> cell update is wave-local.
__global__ __launch_bounds__(256)
void lstm_step_kernel(const __bf16* __restrict__ xb,    // [B,I] bf16, this t
                      const __bf16* __restrict__ h_in,  // [B,H] bf16
                      const __bf16* __restrict__ Wih,   // [4H,I] bf16
                      const __bf16* __restrict__ Whh,   // [4H,H] bf16
                      const float* __restrict__ b_ih,   // [4H]
                      const float* __restrict__ b_hh,   // [4H]
                      float* __restrict__ c,            // [B,H] f32 (in/out)
                      float* __restrict__ out_t,        // d_out + t*B*H
                      __bf16* __restrict__ h_out,       // [B,H] bf16
                      float* __restrict__ h_final,      // last step only
                      float* __restrict__ c_final) {    // last step only
    const int tid     = threadIdx.x;
    const int lane    = tid & 31;
    const int wave    = tid >> 5;
    const int rowBase = wave * 16;
    const int col0    = blockIdx.x * 16;

    // B fragments staged per K-slice: [buf][gate][lane][16 bf16] = 2 x 4 KB.
    __shared__ __align__(32) __bf16 smB[2][4][32][16];

    // This thread's 16-byte chunk of the 4 KB slice (256 threads cover it).
    const int fg    = tid >> 6;          // gate 0..3
    const int fr    = tid & 63;
    const int flane = fr >> 1;           // fragment lane 0..31
    const int fe0   = (fr & 1) * 8;      // element offset 0 or 8
    const int fn    = flane & 15;        // N within tile
    const int fkb   = (flane & 16) ? 16 : 0;

    v8f acc[4] = {v8f{}, v8f{}, v8f{}, v8f{}};

    // Prologue: stage slice 0.
    {
        SliceSrc ss = slice_src(0, xb, h_in, Wih, Whh);
        v8bf v = *reinterpret_cast<const v8bf*>(
            ss.W + (size_t)(fg * H + col0 + fn) * ss.ld + ss.k0 + fkb + fe0);
        *reinterpret_cast<v8bf*>(&smB[0][fg][flane][fe0]) = v;
    }

    for (int s = 0; s < KSLICES; ++s) {
        __syncthreads();  // slice s resident in smB[s&1]

        // Stage slice s+1 into the other buffer while computing slice s.
        if (s + 1 < KSLICES) {
            SliceSrc sn = slice_src(s + 1, xb, h_in, Wih, Whh);
            const __bf16* gp =
                sn.W + (size_t)(fg * H + col0 + fn) * sn.ld + sn.k0 + fkb + fe0;
            v8bf v = *reinterpret_cast<const v8bf*>(gp);
            *reinterpret_cast<v8bf*>(&smB[(s + 1) & 1][fg][flane][fe0]) = v;
            if (s + 2 < KSLICES) {  // prefetch slice s+2 of the weight stream
                SliceSrc sp = slice_src(s + 2, xb, h_in, Wih, Whh);
                __builtin_prefetch(
                    sp.W + (size_t)(fg * H + col0 + fn) * sp.ld + sp.k0 + fkb + fe0,
                    0, 1);
            }
        }

        // Compute: A from global (per-wave rows), B from LDS (shared).
        SliceSrc ss = slice_src(s, xb, h_in, Wih, Whh);
        v16bf a = load_a16x32(ss.A, ss.ld, rowBase, ss.k0, lane);
#pragma unroll
        for (int g = 0; g < 4; ++g) {
            v16bf b = *reinterpret_cast<const v16bf*>(&smB[s & 1][g][lane][0]);
            acc[g] = __builtin_amdgcn_wmma_f32_16x16x32_bf16(
                false, a, false, b, (short)0, acc[g], false, false);
        }
        __syncthreads();  // reads of smB[s&1] done before it is refilled
    }

    // Biases: one column per lane, per gate.
    const int n = lane & 15;
    float bias[4];
#pragma unroll
    for (int g = 0; g < 4; ++g) {
        const int cg = g * H + col0 + n;
        bias[g] = b_ih[cg] + b_hh[cg];
    }

    // Elementwise cell update. C/D layout: elem j -> M=j (lanes 0-15),
    // M=j+8 (lanes 16-31); N = lane&15.
    const int rofs = (lane & 16) ? 8 : 0;
    const int col  = col0 + n;
#pragma unroll
    for (int j = 0; j < 8; ++j) {
        const int    m   = rowBase + rofs + j;
        const size_t idx = (size_t)m * H + col;
        const float ig = sigmoidf_fast(acc[0][j] + bias[0]);
        const float fg_ = sigmoidf_fast(acc[1][j] + bias[1]);
        const float gg = tanhf(acc[2][j] + bias[2]);
        const float og = sigmoidf_fast(acc[3][j] + bias[3]);
        const float c_new = fg_ * c[idx] + ig * gg;
        const float h_new = og * tanhf(c_new);
        c[idx]     = c_new;
        out_t[idx] = h_new;
        h_out[idx] = (__bf16)h_new;
        if (h_final) { h_final[idx] = h_new; c_final[idx] = c_new; }
    }
}

// ---------------------------------------------------------------------------
extern "C" void kernel_launch(void* const* d_in, const int* in_sizes, int n_in,
                              void* d_out, int out_size, void* d_ws, size_t ws_size,
                              hipStream_t stream) {
    (void)in_sizes; (void)n_in; (void)out_size; (void)ws_size;

    const float* inputs = (const float*)d_in[0];   // [T,B,I]
    const float* h0     = (const float*)d_in[1];   // [B,H]
    const float* c0     = (const float*)d_in[2];   // [B,H]
    const float* W_ih   = (const float*)d_in[3];   // [4H,I]
    const float* W_hh   = (const float*)d_in[4];   // [4H,H]
    const float* b_ih   = (const float*)d_in[5];   // [4H]
    const float* b_hh   = (const float*)d_in[6];   // [4H]
    float* out = (float*)d_out;                    // outputs | h_final | c_final

    // Workspace layout:
    char* ws = (char*)d_ws;
    const size_t nX   = (size_t)T * B * I;
    const size_t nWih = (size_t)4 * H * I;
    const size_t nWhh = (size_t)4 * H * H;
    const size_t nBH  = (size_t)B * H;
    __bf16* Xb   = (__bf16*)(ws);
    __bf16* Wihb = (__bf16*)(ws + nX * 2);
    __bf16* Whhb = (__bf16*)(ws + nX * 2 + nWih * 2);
    __bf16* hb0  = (__bf16*)(ws + nX * 2 + nWih * 2 + nWhh * 2);
    __bf16* hb1  = (__bf16*)(ws + nX * 2 + nWih * 2 + nWhh * 2 + nBH * 2);
    float*  cbuf = (float*) (ws + nX * 2 + nWih * 2 + nWhh * 2 + nBH * 4);

    // bf16 conversions + state init (re-done every launch: deterministic).
    f32_to_bf16_kernel<<<(int)((nX   + 255) / 256), 256, 0, stream>>>(inputs, Xb,   (int)nX);
    f32_to_bf16_kernel<<<(int)((nWih + 255) / 256), 256, 0, stream>>>(W_ih,  Wihb, (int)nWih);
    f32_to_bf16_kernel<<<(int)((nWhh + 255) / 256), 256, 0, stream>>>(W_hh,  Whhb, (int)nWhh);
    f32_to_bf16_kernel<<<(int)((nBH  + 255) / 256), 256, 0, stream>>>(h0,    hb0,  (int)nBH);
    hipMemcpyAsync(cbuf, c0, nBH * sizeof(float), hipMemcpyDeviceToDevice, stream);

    float* h_final = out + (size_t)T * B * H;
    float* c_final = h_final + nBH;

    for (int t = 0; t < T; ++t) {
        const __bf16* h_in  = (t & 1) ? hb1 : hb0;
        __bf16*       h_out = (t & 1) ? hb0 : hb1;
        const bool last = (t == T - 1);
        lstm_step_kernel<<<H / 16, 256, 0, stream>>>(
            Xb + (size_t)t * B * I, h_in, Wihb, Whhb, b_ih, b_hh,
            cbuf, out + (size_t)t * B * H, h_out,
            last ? h_final : nullptr, last ? c_final : nullptr);
    }
}
